// TernaryLinear_22402549416019
// MI455X (gfx1250) — matrix operand order repaired and checked by reading, not compiled, not run on hardware
//
#include <hip/hip_runtime.h>
#include <hip/hip_bf16.h>

// ---------------------------------------------------------------------------
// TernaryLinear on gfx1250: out[8192,8192] = x[8192,2048] @ ternary(W)[2048,8192]
// Compute-bound GEMM -> bf16 WMMA (v_wmma_f32_16x16x32_bf16), f32 accumulate.
// Data movement: double-buffered global->LDS via GLOBAL_LOAD_ASYNC_TO_LDS_B128
// (ASYNCcnt), fragments fed with ds_load_b128.
// Software pipeline written so the buffer index is a compile-time constant in
// every stage: branch-free steady-state loop, peeled epilogue.
// ---------------------------------------------------------------------------

typedef __attribute__((ext_vector_type(16))) __bf16 v16bf;
typedef __attribute__((ext_vector_type(8)))  float  v8f;
typedef __attribute__((ext_vector_type(4)))  int    v4i;

union BFrag {            // 8 VGPRs per lane = 16 bf16 elements
    v16bf bf;
    v4i   i4[2];
};
static_assert(sizeof(BFrag) == 32, "frag size");

constexpr int M_DIM = 8192;   // 4 * 2048 rows of x
constexpr int K_DIM = 2048;
constexpr int N_DIM = 8192;
constexpr float THRESH = 0.3f;

// ---------------------------------------------------------------------------
// Kernel 1: ternary-quantize W [K,N] f32 and transpose into Wt [N,K] bf16.
// ---------------------------------------------------------------------------
__global__ __launch_bounds__(256)
void quant_transpose(const float* __restrict__ w, __hip_bfloat16* __restrict__ wt) {
    __shared__ __hip_bfloat16 tile[32][33];
    const int n0 = blockIdx.x * 32;
    const int k0 = blockIdx.y * 32;
    const int tx = threadIdx.x;       // 0..31
    const int ty = threadIdx.y;       // 0..7
#pragma unroll
    for (int r = 0; r < 32; r += 8) {
        float f = w[(size_t)(k0 + ty + r) * N_DIM + (n0 + tx)];
        float t = (f > THRESH) ? 1.0f : ((f < -THRESH) ? -1.0f : 0.0f);
        tile[ty + r][tx] = __float2bfloat16(t);
    }
    __syncthreads();
#pragma unroll
    for (int r = 0; r < 32; r += 8) {
        wt[(size_t)(n0 + ty + r) * K_DIM + (k0 + tx)] = tile[tx][ty + r];
    }
}

// ---------------------------------------------------------------------------
// Kernel 2: cast x f32 -> bf16.
// ---------------------------------------------------------------------------
__global__ __launch_bounds__(256)
void cast_x(const float* __restrict__ x, __hip_bfloat16* __restrict__ xb, int n) {
    int i = (blockIdx.x * blockDim.x + threadIdx.x) * 4;
    if (i + 3 < n) {
        float4 v = *(const float4*)(x + i);
        xb[i + 0] = __float2bfloat16(v.x);
        xb[i + 1] = __float2bfloat16(v.y);
        xb[i + 2] = __float2bfloat16(v.z);
        xb[i + 3] = __float2bfloat16(v.w);
    }
}

// ---------------------------------------------------------------------------
// Async copy helpers (gfx1250 ASYNCcnt path).
// ---------------------------------------------------------------------------
__device__ __forceinline__ void async_copy_b128(const void* lds_dst, const void* gsrc) {
    unsigned int  lds_addr = (unsigned int)(unsigned long long)lds_dst;
    unsigned long long gaddr = (unsigned long long)gsrc;
    asm volatile("global_load_async_to_lds_b128 %0, %1, off"
                 :: "v"(lds_addr), "v"(gaddr)
                 : "memory");
}
__device__ __forceinline__ void async_wait_le8() {
    asm volatile("s_wait_asynccnt 0x8" ::: "memory");
}
__device__ __forceinline__ void async_wait_zero() {
    asm volatile("s_wait_asynccnt 0x0" ::: "memory");
}

// ---------------------------------------------------------------------------
// Kernel 3: GEMM. A = xb [M,K] bf16 row-major, B = Wt [N,K] bf16 (N-major).
//
// Block: 128 threads = 4 waves (2 M x 2 N), block tile 128(M) x 128(N).
// Wave tile: 64(M) x 64(N) = 4x4 accumulators -> 16 WMMA per 32-K step,
// 1 ds_load_b128 per WMMA, A/B fragments reused 4x from registers.
//
// LDS: double-buffered A[128][32] / B[128][32] bf16 tiles, row stride 56
// elements (112 B): 16-B aligned ds_load_b128, and row*28 dwords mod 64 is a
// permutation over 16 rows -> conflict-free fragment reads. 56 KB total.
//
// Pipeline (buf constant in every stage):
//   issue(0, k=0)
//   loop k0 += 64 (rolled):
//     issue(1, k0+32); wait<=8; bar; mma(buf0); bar;
//     issue(0, k0+64); wait<=8; bar; mma(buf1); bar;
//   epilogue: issue(1, K-32); wait<=8; bar; mma(buf0); bar;
//             wait 0;                  bar; mma(buf1)
// Async loads of a wave retire in order, so wait<=8 after issuing the next
// buffer's 8 copies proves the current buffer's 8 have landed; the barrier
// extends that block-wide.
// ---------------------------------------------------------------------------
constexpr int LSTR = 56;   // padded LDS row stride in bf16 elements (112 B)

__global__ __launch_bounds__(128)
void ternary_gemm(const __hip_bfloat16* __restrict__ A,
                  const __hip_bfloat16* __restrict__ B,
                  float* __restrict__ C) {
    __shared__ __align__(16) __hip_bfloat16 tA[2][128][LSTR];
    __shared__ __align__(16) __hip_bfloat16 tB[2][128][LSTR];

    const int tid    = threadIdx.x;          // 0..127
    const int lane   = tid & 31;
    const int wave   = tid >> 5;             // 0..3
    const int laneLo = lane & 15;
    const int laneHi = lane >> 4;            // 0 or 1

    const int m_blk = blockIdx.y * 128;
    const int n_blk = blockIdx.x * 128;
    const int wm    = (wave & 1) * 64;       // wave tile M offset in block
    const int wn    = (wave >> 1) * 64;      // wave tile N offset in block

    // Copy assignment: each tile = 128 rows x 4 chunks of 16 B = 512 chunks;
    // 4 chunks per thread per matrix (chunk ids tid + 128*q).
    int rr[4], ss[4];
#pragma unroll
    for (int q = 0; q < 4; ++q) {
        int c = tid + 128 * q;
        rr[q] = c >> 2;
        ss[q] = (c & 3) * 8;                 // element offset within row
    }

    const __hip_bfloat16* gA = A + (size_t)(m_blk)*K_DIM;
    const __hip_bfloat16* gB = B + (size_t)(n_blk)*K_DIM;

    v8f acc[4][4] = {};

    auto issue_tile = [&](int buf, int k0) {
#pragma unroll
        for (int q = 0; q < 4; ++q)
            async_copy_b128(&tA[buf][rr[q]][ss[q]],
                            gA + (size_t)rr[q] * K_DIM + k0 + ss[q]);
#pragma unroll
        for (int q = 0; q < 4; ++q)
            async_copy_b128(&tB[buf][rr[q]][ss[q]],
                            gB + (size_t)rr[q] * K_DIM + k0 + ss[q]);
    };

    auto mma_tile = [&](int buf) {
        BFrag a[4], b[4];
#pragma unroll
        for (int i = 0; i < 4; ++i) {
            const __hip_bfloat16* p = &tA[buf][wm + i * 16 + laneLo][laneHi * 8];
            a[i].i4[0] = *(const v4i*)(p);        // K = +0..7  (per half-wave)
            a[i].i4[1] = *(const v4i*)(p + 16);   // K = +16..23
        }
#pragma unroll
        for (int j = 0; j < 4; ++j) {
            const __hip_bfloat16* p = &tB[buf][wn + j * 16 + laneLo][laneHi * 8];
            b[j].i4[0] = *(const v4i*)(p);        // K = +0..7  (per half-wave)
            b[j].i4[1] = *(const v4i*)(p + 16);   // K = +16..23
        }
#pragma unroll
        for (int i = 0; i < 4; ++i) {
#pragma unroll
            for (int j = 0; j < 4; ++j) {
                acc[i][j] = __builtin_amdgcn_wmma_f32_16x16x32_bf16(
                    false, a[i].bf, false, b[j].bf,
                    (short)0, acc[i][j], false, false);
            }
        }
    };

    issue_tile(0, 0);

    // Steady state: branch-free, buf indices are literals in both stages.
#pragma unroll 1
    for (int k0 = 0; k0 < K_DIM - 64; k0 += 64) {
        issue_tile(1, k0 + 32);
        async_wait_le8();
        __syncthreads();
        mma_tile(0);
        __syncthreads();

        issue_tile(0, k0 + 64);
        async_wait_le8();
        __syncthreads();
        mma_tile(1);
        __syncthreads();
    }

    // Epilogue: slabs K-64 (in buf0) and K-32.
    issue_tile(1, K_DIM - 32);
    async_wait_le8();
    __syncthreads();
    mma_tile(0);
    __syncthreads();

    async_wait_zero();
    __syncthreads();
    mma_tile(1);

#pragma unroll
    for (int i = 0; i < 4; ++i) {
#pragma unroll
        for (int j = 0; j < 4; ++j) {
            const int row = m_blk + wm + i * 16 + laneHi * 8;
            const int col = n_blk + wn + j * 16 + laneLo;
#pragma unroll
            for (int r = 0; r < 8; ++r) {
                C[(size_t)(row + r) * N_DIM + col] = acc[i][j][r];
            }
        }
    }
}

// ---------------------------------------------------------------------------
extern "C" void kernel_launch(void* const* d_in, const int* in_sizes, int n_in,
                              void* d_out, int out_size, void* d_ws, size_t ws_size,
                              hipStream_t stream) {
    const float* x = (const float*)d_in[0];   // [4,2048,2048] f32
    const float* w = (const float*)d_in[1];   // [2048,8192]   f32
    float* out = (float*)d_out;               // [4,2048,8192] f32

    // Workspace: Wt bf16 [N,K] = 32 MiB, then xb bf16 [M,K] = 32 MiB.
    __hip_bfloat16* wt = (__hip_bfloat16*)d_ws;
    __hip_bfloat16* xb =
        (__hip_bfloat16*)((char*)d_ws + (size_t)N_DIM * K_DIM * sizeof(__hip_bfloat16));

    dim3 bq(32, 8);
    dim3 gq(N_DIM / 32, K_DIM / 32);
    quant_transpose<<<gq, bq, 0, stream>>>(w, wt);

    const int nx = M_DIM * K_DIM;
    cast_x<<<(nx / 4 + 255) / 256, 256, 0, stream>>>(x, xb, nx);

    dim3 gg(N_DIM / 128, M_DIM / 128);
    ternary_gemm<<<gg, 128, 0, stream>>>(xb, wt, out);
}
